// EncoderStack_57939108823211
// MI455X (gfx1250) — compile-verified, matrix operand order checked by reading, not run on hardware
//
#include <hip/hip_runtime.h>
#include <math.h>
#include <stdint.h>

// ---------------------------------------------------------------------------
// Types for CDNA5 WMMA
// ---------------------------------------------------------------------------
typedef __attribute__((ext_vector_type(16))) _Float16 v16h;
typedef __attribute__((ext_vector_type(8)))  _Float16 h8;
typedef __attribute__((ext_vector_type(4)))  _Float16 h4;
typedef __attribute__((ext_vector_type(8)))  float    v8f;
typedef __attribute__((ext_vector_type(4)))  float    f32x4;

// ---------------------------------------------------------------------------
// Problem constants
// ---------------------------------------------------------------------------
#define BC      112          // batch*n_vars
#define NP      32           // num patches
#define PATCH   16
#define DM      128          // d_model
#define SEQ     510
#define PD      2048         // PATCH*DM
#define DFF     512
#define ROWS_BN 3584         // BC*NP
#define ROWS_HP 57344        // BC*NP*PATCH

// ===========================================================================
// Generic WMMA GEMM:  C[M,N] = A[M,K] @ B[K,N]  (+bias +peA +peC +res, relu)
//   - fp32 in/out, f16 WMMA compute, fp32 accumulate
//   - block = 256 threads (8 wave32), 128x128 tile, BK=32
//   - double-buffered LDS, software-pipelined: one barrier per K-step,
//     next-tile global loads issued before the 8 v_wmma of the current tile
//   - B staged transposed with 4x4 register sub-block -> 4 packed ds_store_b64
//   - peA:  A_eff[r][k] = A[r][k] + peA[(r % peA_mod)*K + k]      (pre-add PE)
//   - peC:  C += peC[(r & 31)*128 + (c & 127)]                     (PE_INTER)
//   - res:  C += res[r*N + c]   (res may alias C: read-before-write per elem)
// ===========================================================================
__global__ __launch_bounds__(256)
void gemm_wmma(const float* __restrict__ A, const float* __restrict__ B,
               const float* __restrict__ bias, float* C,
               int M, int N, int K,
               const float* __restrict__ peA, int peA_mod,
               const float* __restrict__ peC,
               const float* res,
               int relu)
{
    __shared__ __align__(16) _Float16 lA[2 * 128 * 32];   // [buf][row][k]
    __shared__ __align__(16) _Float16 lB[2 * 128 * 32];   // [buf][col][k]

    const int tid  = threadIdx.x;
    const int lane = tid & 31;
    const int wave = tid >> 5;
    const int wm   = (wave >> 1) * 32;   // wave row offset inside tile
    const int wn   = (wave & 1) * 64;    // wave col offset inside tile
    const int bm   = blockIdx.y * 128;
    const int bn   = blockIdx.x * 128;

    const int g  = lane >> 4;            // half-wave group (K-half select)
    const int rr = lane & 15;            // row/col within 16

    // staging coordinates
    const int ar  = tid >> 3;                // A row slice start (uses +32*it)
    const int ac4 = (tid & 7) << 2;          // A col within BK
    const int bkq = (tid >> 5) << 2;         // B k-row start (4 consecutive rows)
    const int bc4 = (tid & 31) << 2;         // B col start (4 consecutive cols)

    auto loadA = [&](int k0, f32x4* regs) {
#pragma unroll
        for (int it = 0; it < 4; ++it) {
            int r = ar + it * 32;
            f32x4 av = *(const f32x4*)(A + (size_t)(bm + r) * K + k0 + ac4);
            if (peA) {
                const float* pp = peA + (size_t)((bm + r) % peA_mod) * K + k0 + ac4;
                av.x += pp[0]; av.y += pp[1]; av.z += pp[2]; av.w += pp[3];
            }
            regs[it] = av;
        }
    };
    auto loadB = [&](int k0, f32x4* regs) {
#pragma unroll
        for (int it = 0; it < 4; ++it)       // 4 consecutive k-rows, same cols
            regs[it] = *(const f32x4*)(B + (size_t)(k0 + bkq + it) * N + bn + bc4);
    };
    auto storeA = [&](int buf, const f32x4* regs) {
        _Float16* d = lA + buf * 4096;
#pragma unroll
        for (int it = 0; it < 4; ++it) {
            int r = ar + it * 32;
            h4 hv;
            hv.x = (_Float16)regs[it].x; hv.y = (_Float16)regs[it].y;
            hv.z = (_Float16)regs[it].z; hv.w = (_Float16)regs[it].w;
            *(h4*)(d + r * 32 + ac4) = hv;
        }
    };
    auto storeB = [&](int buf, const f32x4* regs) {
        _Float16* d = lB + buf * 4096;
#pragma unroll
        for (int j = 0; j < 4; ++j) {        // per column: 4 consecutive k -> b64
            h4 hv;
            hv.x = (_Float16)regs[0][j];
            hv.y = (_Float16)regs[1][j];
            hv.z = (_Float16)regs[2][j];
            hv.w = (_Float16)regs[3][j];
            *(h4*)(d + (bc4 + j) * 32 + bkq) = hv;
        }
    };

    v8f zero = {};
    v8f acc[2][4];
#pragma unroll
    for (int i = 0; i < 2; ++i)
#pragma unroll
        for (int j = 0; j < 4; ++j) acc[i][j] = zero;

    // ---- prologue: stage tile 0 into buffer 0 ----
    {
        f32x4 ra[4], rb[4];
        loadA(0, ra); loadB(0, rb);
        storeA(0, ra); storeB(0, rb);
    }
    __syncthreads();

    int cur = 0;
    for (int k0 = 0; k0 < K; k0 += 32) {
        const int nk = k0 + 32;
        const bool has_next = nk < K;
        f32x4 ra[4], rb[4];
        if (has_next) {             // issue next-tile global loads early
            loadA(nk, ra);
            loadB(nk, rb);
        }

        // ---- fragments from LDS buffer `cur` + 8 WMMAs ----
        const _Float16* pAb = lA + cur * 4096;
        const _Float16* pBb = lB + cur * 4096;
        v16h af[2], bf[4];
#pragma unroll
        for (int i = 0; i < 2; ++i) {
            const _Float16* p = pAb + (wm + i * 16 + rr) * 32;
            h8 lo = *(const h8*)(p + g * 8);
            h8 hi = *(const h8*)(p + 16 + g * 8);
#pragma unroll
            for (int e = 0; e < 8; ++e) { af[i][e] = lo[e]; af[i][8 + e] = hi[e]; }
        }
#pragma unroll
        for (int j = 0; j < 4; ++j) {
            const _Float16* p = pBb + (wn + j * 16 + rr) * 32;
            h8 lo = *(const h8*)(p + g * 8);
            h8 hi = *(const h8*)(p + 16 + g * 8);
#pragma unroll
            for (int e = 0; e < 8; ++e) { bf[j][e] = lo[e]; bf[j][8 + e] = hi[e]; }
        }
#pragma unroll
        for (int i = 0; i < 2; ++i)
#pragma unroll
            for (int j = 0; j < 4; ++j)
                acc[i][j] = __builtin_amdgcn_wmma_f32_16x16x32_f16(
                    false, af[i], false, bf[j], (short)0, acc[i][j], false, false);

        if (has_next) {             // convert + store into the other buffer
            storeA(cur ^ 1, ra);
            storeB(cur ^ 1, rb);
        }
        __syncthreads();
        cur ^= 1;
    }

    // ---- epilogue: bias / residual / PE_INTER / relu ----
#pragma unroll
    for (int i = 0; i < 2; ++i) {
#pragma unroll
        for (int j = 0; j < 4; ++j) {
            int col = bn + wn + j * 16 + rr;
            float bb = bias ? bias[col] : 0.f;
#pragma unroll
            for (int v = 0; v < 8; ++v) {
                int row = bm + wm + i * 16 + g * 8 + v;
                size_t off = (size_t)row * N + col;
                float o = acc[i][j][v] + bb;
                if (res) o += res[off];
                if (peC) o += peC[(row & 31) * 128 + (col & 127)];
                if (relu) o = o > 0.f ? o : 0.f;
                C[off] = o;
            }
        }
    }
}

// ===========================================================================
// Positional encoding table: sincos PE, zero-mean, /(std(ddof=1)*10)
// ===========================================================================
__device__ inline double pe_raw(int e)
{
    int pos = e >> 7, i = e & 127;
    int half = i >> 1;
    double div = exp((double)(2 * half) * (-log(10000.0) / 128.0));
    double a = (double)pos * div;
    return (i & 1) ? cos(a) : sin(a);
}

__global__ void pe_kernel(float* __restrict__ dst, int qlen)
{
    __shared__ double red[256], red2[256];
    __shared__ double s_mean, s_inv;
    const int n = qlen * 128;
    double sum = 0.0, sq = 0.0;
    for (int e = threadIdx.x; e < n; e += 256) {
        double v = pe_raw(e);
        sum += v; sq += v * v;
    }
    red[threadIdx.x] = sum; red2[threadIdx.x] = sq;
    __syncthreads();
    if (threadIdx.x == 0) {
        double S = 0.0, Q = 0.0;
        for (int i = 0; i < 256; ++i) { S += red[i]; Q += red2[i]; }
        double mean = S / n;
        double var  = (Q - n * mean * mean) / (double)(n - 1);
        s_mean = mean;
        s_inv  = 1.0 / (sqrt(var) * 10.0);
    }
    __syncthreads();
    for (int e = threadIdx.x; e < n; e += 256)
        dst[e] = (float)((pe_raw(e) - s_mean) * s_inv);
}

// ===========================================================================
// Patchify (replicate-pad 510 -> 512) and crop-back
// ===========================================================================
__global__ void patchify(const float* __restrict__ X, float* __restrict__ H)
{
    int idx = blockIdx.x * 256 + threadIdx.x;        // float4 index
    if (idx >= BC * 512 * 32) return;
    int q = idx & 31; int rest = idx >> 5;
    int s = rest & 511; int b = rest >> 9;
    int ss = s < SEQ ? s : SEQ - 1;
    f32x4 v = *(const f32x4*)(X + ((size_t)b * SEQ + ss) * DM + q * 4);
    *(f32x4*)(H + (size_t)idx * 4) = v;
}

__global__ void crop(const float* __restrict__ H, float* __restrict__ OUT)
{
    int idx = blockIdx.x * 256 + threadIdx.x;        // float4 index
    if (idx >= BC * SEQ * 32) return;
    int q = idx & 31; int rest = idx >> 5;
    int t = rest % SEQ; int b = rest / SEQ;
    f32x4 v = *(const f32x4*)(H + ((size_t)b * 512 + t) * DM + q * 4);
    *(f32x4*)(OUT + (size_t)idx * 4) = v;
}

// ===========================================================================
// LayerNorm over last dim (128): one wave32 per row, 8 rows per block
// ===========================================================================
__global__ __launch_bounds__(256)
void layernorm_kernel(const float* __restrict__ src, const float* __restrict__ res,
                      const float* __restrict__ g, const float* __restrict__ b,
                      float* __restrict__ dst, int rows)
{
    int wave = threadIdx.x >> 5, lane = threadIdx.x & 31;
    int row = blockIdx.x * 8 + wave;
    if (row >= rows) return;
    const size_t base = (size_t)row * 128;
    float x[4];
#pragma unroll
    for (int i = 0; i < 4; ++i) {
        int c = lane + i * 32;
        x[i] = src[base + c] + (res ? res[base + c] : 0.f);
    }
    float sum = x[0] + x[1] + x[2] + x[3];
#pragma unroll
    for (int o = 16; o > 0; o >>= 1) sum += __shfl_xor(sum, o, 32);
    float mean = sum * (1.f / 128.f);
    float vs = 0.f;
#pragma unroll
    for (int i = 0; i < 4; ++i) { float d = x[i] - mean; vs += d * d; }
#pragma unroll
    for (int o = 16; o > 0; o >>= 1) vs += __shfl_xor(vs, o, 32);
    float inv = rsqrtf(vs * (1.f / 128.f) + 1e-5f);
#pragma unroll
    for (int i = 0; i < 4; ++i) {
        int c = lane + i * 32;
        dst[base + c] = (x[i] - mean) * inv * g[c] + b[c];
    }
}

// ===========================================================================
// Token embedding: TOKE[b*32+n, d] = sum_k tok[b%7,n,k]*emb_W[n,k,d] + emb_b[n,d]
// ===========================================================================
__global__ __launch_bounds__(128)
void tok_emb(const float* __restrict__ tok, const float* __restrict__ embW,
             const float* __restrict__ embb, float* __restrict__ TOKE)
{
    int bn = blockIdx.x, d = threadIdx.x;
    int b = bn >> 5, n = bn & 31, v = b % 7;
    float acc = embb[n * 128 + d];
#pragma unroll
    for (int k = 0; k < 16; ++k)
        acc += tok[(v * 32 + n) * 16 + k] * embW[(n * 16 + k) * 128 + d];
    TOKE[(size_t)bn * 128 + d] = acc;
}

// ===========================================================================
// Intra-periodic attention core: per (b,n): q[128] attends over 16 patch rows
// ===========================================================================
__global__ __launch_bounds__(128)
void intra_attn(const float* __restrict__ Q, const float* __restrict__ K,
                const float* __restrict__ V, float* __restrict__ O)
{
    __shared__ float q[128];
    __shared__ float kk[16 * 128];
    __shared__ float vv[16 * 128];
    __shared__ float s[128];
    int bn = blockIdx.x, t = threadIdx.x;
    q[t] = Q[(size_t)bn * 128 + t];
    for (int p = 0; p < 16; ++p) {
        kk[p * 128 + t] = K[((size_t)bn * 16 + p) * 128 + t];
        vv[p * 128 + t] = V[((size_t)bn * 16 + p) * 128 + t];
    }
    __syncthreads();
    int h = t >> 4, p = t & 15;
    float acc = 0.f;
#pragma unroll
    for (int dd = 0; dd < 16; ++dd)
        acc += q[h * 16 + dd] * kk[p * 128 + h * 16 + dd];
    s[h * 16 + p] = acc * 0.25f;                  // scale = 1/sqrt(16)
    __syncthreads();
    float mx = -1e30f;
    for (int pp = 0; pp < 16; ++pp) mx = fmaxf(mx, s[h * 16 + pp]);
    float sm = 0.f;
    for (int pp = 0; pp < 16; ++pp) sm += expf(s[h * 16 + pp] - mx);
    float inv = 1.f / sm;
    float o = 0.f;
    for (int pp = 0; pp < 16; ++pp)
        o += expf(s[h * 16 + pp] - mx) * inv * vv[pp * 128 + t];
    O[(size_t)bn * 128 + t] = o;
}

// ===========================================================================
// Intra out-proj: Linear(N=32 -> N*P=512) along patch axis, transposed store
// ===========================================================================
__global__ __launch_bounds__(128)
void intra_proj(const float* __restrict__ O, const float* __restrict__ ioW,
                const float* __restrict__ iob, float* __restrict__ ATT)
{
    int m = blockIdx.x, b = blockIdx.y, d = threadIdx.x;
    float acc = iob[m];
#pragma unroll
    for (int n = 0; n < 32; ++n)
        acc += O[((size_t)b * 32 + n) * 128 + d] * ioW[n * 512 + m];
    int n = m >> 4, p = m & 15;
    ATT[(((size_t)b * 32 + n) * 16 + p) * 128 + d] = acc;
}

// ===========================================================================
// Inter-periodic attention core: per (b,h): 32x32 softmax(QK^T*0.25) @ V
// Q/K/V head slices staged to LDS with gfx1250 async global->LDS DMA
// (global_load_async_to_lds_b128 + s_wait_asynccnt), fp32 in dynamic LDS.
// dyn-LDS = 3*32*256*4 + 32*32*4 = 102400 B
// ===========================================================================
__global__ __launch_bounds__(256)
void inter_attn(const float* __restrict__ Q2, const float* __restrict__ K2,
                const float* __restrict__ V2, float* __restrict__ O2)
{
    extern __shared__ float smem[];
    float* Qs = smem;                 // [32][256]
    float* Ks = smem + 32 * 256;
    float* Vs = smem + 2 * 32 * 256;
    float* S  = smem + 3 * 32 * 256;  // [32][32]

    int h = blockIdx.x, b = blockIdx.y, t = threadIdx.x;
    size_t base = (size_t)b * 32 * 2048 + (size_t)h * 256;

    // LDS byte offsets (low 32 bits of flat shared address = LDS offset)
    uint32_t ldsQ = (uint32_t)(uintptr_t)Qs;
    uint32_t ldsK = (uint32_t)(uintptr_t)Ks;
    uint32_t ldsV = (uint32_t)(uintptr_t)Vs;

    // 32 rows x 256 floats = 2048 16-byte chunks per matrix; 8 per thread
#pragma unroll
    for (int it = 0; it < 8; ++it) {
        int chunk = t + it * 256;          // 0..2047
        int i   = chunk >> 6;              // row 0..31
        int o16 = chunk & 63;              // 16B chunk in row
        size_t goff = base + (size_t)i * 2048 + o16 * 4;
        uint32_t l = (uint32_t)((i * 256 + o16 * 4) * 4);
        const float* gq = Q2 + goff;
        const float* gk = K2 + goff;
        const float* gv = V2 + goff;
        asm volatile("global_load_async_to_lds_b128 %0, %1, off"
                     :: "v"(ldsQ + l), "v"(gq) : "memory");
        asm volatile("global_load_async_to_lds_b128 %0, %1, off"
                     :: "v"(ldsK + l), "v"(gk) : "memory");
        asm volatile("global_load_async_to_lds_b128 %0, %1, off"
                     :: "v"(ldsV + l), "v"(gv) : "memory");
    }
    asm volatile("s_wait_asynccnt 0x0" ::: "memory");
    __syncthreads();

    int i = t >> 3;
#pragma unroll
    for (int sj = 0; sj < 4; ++sj) {
        int j = (t & 7) * 4 + sj;
        const float* qp = Qs + i * 256;
        const float* kp = Ks + j * 256;
        float acc = 0.f;
        for (int d = 0; d < 256; ++d) acc += qp[d] * kp[d];
        S[i * 32 + j] = acc * 0.25f;              // source scale 1/sqrt(16)
    }
    __syncthreads();
    if (t < 32) {
        float mx = -1e30f;
        for (int j = 0; j < 32; ++j) mx = fmaxf(mx, S[t * 32 + j]);
        float sm = 0.f;
        for (int j = 0; j < 32; ++j) { float e = expf(S[t * 32 + j] - mx); S[t * 32 + j] = e; sm += e; }
        float inv = 1.f / sm;
        for (int j = 0; j < 32; ++j) S[t * 32 + j] *= inv;
    }
    __syncthreads();
    int dbase = (t & 7) * 32;
    for (int dd = 0; dd < 32; ++dd) {
        int d = dbase + dd;
        float acc = 0.f;
        for (int j = 0; j < 32; ++j) acc += S[i * 32 + j] * Vs[j * 256 + d];
        O2[base + (size_t)i * 2048 + d] = acc;
    }
}

// ===========================================================================
// Host-side orchestration
// ===========================================================================
static inline void launch_gemm(const float* A, const float* B, const float* bias,
                               float* C, int M, int N, int K,
                               const float* peA, int peA_mod,
                               const float* peC, const float* res, int relu,
                               hipStream_t stream)
{
    dim3 g(N / 128, M / 128);
    gemm_wmma<<<g, 256, 0, stream>>>(A, B, bias, C, M, N, K,
                                     peA, peA_mod > 0 ? peA_mod : 1, peC, res, relu);
}

extern "C" void kernel_launch(void* const* d_in, const int* in_sizes, int n_in,
                              void* d_out, int out_size, void* d_ws, size_t ws_size,
                              hipStream_t stream)
{
    (void)in_sizes; (void)n_in; (void)out_size; (void)ws_size;
    const float* X = (const float*)d_in[0];
    // Per-layer params (recursive dict-insertion-order flattening):
    // tok,emb_W,emb_b,iq_W,iq_b,ik_W,ik_b,iv_W,iv_b,io_W,io_b,e_W,e_b,eq_W,eq_b,
    // ek_W,ek_b,ev_W,ev_b,eo_W,eo_b,ln1_g,ln1_b,ln2_g,ln2_b,f1_W,f1_b,f2_W,f2_b
    auto P = [&](int l, int k) { return (const float*)d_in[1 + l * 29 + k]; };
    const float* fn_g = (const float*)d_in[1 + 2 * 29 + 0];
    const float* fn_b = (const float*)d_in[1 + 2 * 29 + 1];

    // ---- workspace layout (floats). SZ = 3584*2048 = 57344*128 ----
    const size_t SZ = 7340032ull;
    float* ws  = (float*)d_ws;
    float* H   = ws;              // hidden state [112,32,16,128]
    float* B1  = ws + 1 * SZ;     // XR ; also MID[57344,512] spans B1..B4
    float* B2  = ws + 2 * SZ;     // intra-K, then Q2
    float* B3  = ws + 3 * SZ;     // intra-V, then K2
    float* B4  = ws + 4 * SZ;     // V2
    float* B5  = ws + 5 * SZ;     // O2
    float* B6  = ws + 6 * SZ;     // intra+inter attn out / ffn out
    float* TOKE = ws + 7 * SZ;            // 3584*128
    float* Qb   = TOKE + 458752;          // 3584*128
    float* Ob   = Qb   + 458752;          // 3584*128
    float* PEQ  = Ob   + 458752;          // 1*128
    float* PEKV = PEQ  + 128;             // 16*128
    float* PEIN = PEKV + 2048;            // 32*128

    pe_kernel<<<1, 256, 0, stream>>>(PEQ, 1);
    pe_kernel<<<1, 256, 0, stream>>>(PEKV, 16);
    pe_kernel<<<1, 256, 0, stream>>>(PEIN, 32);

    patchify<<<(BC * 512 * 32 + 255) / 256, 256, 0, stream>>>(X, H);

    const size_t INTER_LDS = (3 * 32 * 256 + 32 * 32) * sizeof(float); // 102400 B

    for (int l = 0; l < 2; ++l) {
        const float *tok = P(l,0), *embW = P(l,1), *embb = P(l,2);
        const float *iqW = P(l,3), *iqb = P(l,4), *ikW = P(l,5), *ikb = P(l,6);
        const float *ivW = P(l,7), *ivb = P(l,8), *ioW = P(l,9), *iob = P(l,10);
        const float *eW  = P(l,11), *eb  = P(l,12);
        const float *eqW = P(l,13), *eqb = P(l,14);
        const float *ekW = P(l,15), *ekb = P(l,16);
        const float *evW = P(l,17), *evb = P(l,18);
        const float *eoW = P(l,19), *eob = P(l,20);
        const float *l1g = P(l,21), *l1b = P(l,22);
        const float *l2g = P(l,23), *l2b = P(l,24);
        const float *f1W = P(l,25), *f1b = P(l,26);
        const float *f2W = P(l,27), *f2b = P(l,28);

        // ---- intra-periodic branch ----
        tok_emb<<<ROWS_BN, 128, 0, stream>>>(tok, embW, embb, TOKE);
        launch_gemm(TOKE, iqW, iqb, Qb, ROWS_BN, 128, 128, PEQ, 1, nullptr, nullptr, 0, stream);
        launch_gemm(H,    ikW, ikb, B2, ROWS_HP, 128, 128, PEKV, 16, nullptr, nullptr, 0, stream);
        launch_gemm(H,    ivW, ivb, B3, ROWS_HP, 128, 128, PEKV, 16, nullptr, nullptr, 0, stream);
        intra_attn<<<ROWS_BN, 128, 0, stream>>>(Qb, B2, B3, Ob);
        intra_proj<<<dim3(512, BC), 128, 0, stream>>>(Ob, ioW, iob, B6);

        // ---- inter-periodic branch (WMMA-heavy 2048x2048 GEMMs) ----
        launch_gemm(H,  eW,  eb,  B1, ROWS_BN, PD, PD, nullptr, 1, PEIN, nullptr, 0, stream);
        launch_gemm(B1, eqW, eqb, B2, ROWS_BN, PD, PD, nullptr, 1, nullptr, nullptr, 0, stream);
        launch_gemm(B1, ekW, ekb, B3, ROWS_BN, PD, PD, nullptr, 1, nullptr, nullptr, 0, stream);
        launch_gemm(B1, evW, evb, B4, ROWS_BN, PD, PD, nullptr, 1, nullptr, nullptr, 0, stream);
        inter_attn<<<dim3(8, BC), 256, INTER_LDS, stream>>>(B2, B3, B4, B5);
        launch_gemm(B5, eoW, eob, B6, ROWS_BN, PD, PD, nullptr, 1, nullptr, B6, 0, stream);

        // ---- post-norm + FFN ----
        layernorm_kernel<<<ROWS_HP / 8, 256, 0, stream>>>(B6, H, l1g, l1b, H, ROWS_HP);
        launch_gemm(H,  f1W, f1b, B1, ROWS_HP, DFF, 128, nullptr, 1, nullptr, nullptr, 1, stream);
        launch_gemm(B1, f2W, f2b, B6, ROWS_HP, 128, DFF, nullptr, 1, nullptr, H, 0, stream);
        layernorm_kernel<<<ROWS_HP / 8, 256, 0, stream>>>(B6, nullptr, l2g, l2b, H, ROWS_HP);
    }

    layernorm_kernel<<<ROWS_HP / 8, 256, 0, stream>>>(H, nullptr, fn_g, fn_b, H, ROWS_HP);
    crop<<<(BC * SEQ * 32 + 255) / 256, 256, 0, stream>>>(H, (float*)d_out);
}
